// AdaptiveMixing_12446815224516
// MI455X (gfx1250) — matrix operand order repaired and checked by reading, not compiled, not run on hardware
//
#include <hip/hip_runtime.h>
#include <cstdint>

typedef __bf16 bf16_t;
typedef __attribute__((ext_vector_type(16))) __bf16 v16bf;
typedef __attribute__((ext_vector_type(8)))  float  v8f;

union FragU { uint32_t u[8]; v16bf v; };

__device__ __forceinline__ uint16_t f2bf(float f) {
  union { float f; uint32_t u; } c; c.f = f;
  uint32_t u = c.u;
  uint32_t r = (u + 0x7FFFu + ((u >> 16) & 1u)) >> 16;
  return (uint16_t)r;
}

// A-matrix fragment (16x32 bf16 tile): lane holds row m = lane%16, h = lane/16.
// VGPR i(0..3): K = kbase + h*8 + 2i ; VGPR i(4..7): K = kbase + 16 + h*8 + 2(i-4)
// row: pointer to row as uint32 (2 bf16 per uint); kU = kbase/2
__device__ __forceinline__ v16bf load_frag_A(const uint32_t* row, int kU, int h) {
  FragU f;
#pragma unroll
  for (int i = 0; i < 4; i++) f.u[i] = row[kU + h * 4 + i];
#pragma unroll
  for (int i = 0; i < 4; i++) f.u[4 + i] = row[kU + 8 + h * 4 + i];
  return f.v;
}

// B-matrix fragment (32x16 bf16 tile), stored K-contiguous per column (B^T rows).
// lane holds col nn = lane%16, h = lane/16. VGPR j(0..3): K = kbase + h*8 + 2j ; VGPR j(4..7): K = kbase + 16 + h*8 + 2(j-4)
__device__ __forceinline__ v16bf load_frag_B(const uint32_t* row, int kU, int h) {
  FragU f;
#pragma unroll
  for (int j = 0; j < 8; j++) f.u[j] = row[kU + (j >> 2) * 8 + h * 4 + (j & 3)];
  return f.v;
}

__device__ __forceinline__ v8f wmma_bf16(v16bf a, v16bf b, v8f c) {
  return __builtin_amdgcn_wmma_f32_16x16x32_bf16(false, a, false, b, (short)0, c,
                                                 false, false);
}

// ---------------- K0: transpose + f32->bf16 convert.  src[R][C] f32 -> dst[C][R] bf16
__global__ __launch_bounds__(256) void k0_transpose_cvt(const float* __restrict__ src,
                                                        uint16_t* __restrict__ dst,
                                                        int R, int C) {
  __shared__ float tile[32][33];
  int tx = threadIdx.x & 31;
  int ty = threadIdx.x >> 5;  // 0..7
  int c0 = blockIdx.x * 32;
  int r0 = blockIdx.y * 32;
#pragma unroll
  for (int i = 0; i < 4; i++) {
    int r = r0 + ty + i * 8;
    tile[ty + i * 8][tx] = src[(size_t)r * C + c0 + tx];
  }
  __syncthreads();
#pragma unroll
  for (int i = 0; i < 4; i++) {
    int cc = ty + i * 8;
    dst[(size_t)(c0 + cc) * R + r0 + tx] = f2bf(tile[tx][cc]);
  }
}

// ---------------- K0c: elementwise f32->bf16
__global__ __launch_bounds__(256) void k0_cvt(const float* __restrict__ src,
                                              uint16_t* __restrict__ dst, int n) {
  int i = blockIdx.x * 256 + threadIdx.x;
  if (i < n) dst[i] = f2bf(src[i]);
}

// ---------------- K1: params = q(3600x256) @ Wp(256x32768) + bp  -> bf16 pbuf
// block: 256 thr (8 waves), tile 16 rows x 128 cols. grid (256, 225)
__global__ __launch_bounds__(256) void k1_params(const uint32_t* __restrict__ qbf,
                                                 const uint32_t* __restrict__ WpT,
                                                 const float* __restrict__ bp,
                                                 uint16_t* __restrict__ pbuf) {
  __shared__ uint32_t Qs[16 * 128];  // 16 rows x 256 bf16
  int tid = threadIdx.x;
  int n0 = blockIdx.y * 16;
  int c0 = blockIdx.x * 128;
  for (int i = tid; i < 16 * 128; i += 256) {
    int r = i >> 7, col = i & 127;
    Qs[i] = qbf[(size_t)(n0 + r) * 128 + col];
  }
  __syncthreads();
  int w = tid >> 5, L = tid & 31, h = L >> 4, ln = L & 15;
  int c = c0 + w * 16 + ln;
  const uint32_t* brow = WpT + (size_t)c * 128;  // 256 bf16 K-contiguous
  const uint32_t* arow = Qs + ln * 128;
  v8f acc = {};
#pragma unroll
  for (int kb = 0; kb < 8; kb++) {
    v16bf a = load_frag_A(arow, kb * 16, h);
    v16bf b = load_frag_B(brow, kb * 16, h);
    acc = wmma_bf16(a, b, acc);
  }
  float bpv = bp[c];
#pragma unroll
  for (int r = 0; r < 8; r++) {
    int m = r + 8 * h;
    pbuf[(size_t)(n0 + m) * 32768 + c] = f2bf(acc[r] + bpv);
  }
}

// ---------------- K2: per (n,g) adaptive mixing, fused LN+ReLU, writes bf16 over params
// block: 128 thr (4 waves). grid (3600, 4)
__global__ __launch_bounds__(128) void k2_mix(const float* __restrict__ x,
                                              uint16_t* __restrict__ pbuf) {
  __shared__ uint16_t Xsh[32 * 64];    // A of step3, row-major [p][c]
  __shared__ uint16_t MshT[64 * 64];   // B of step3 transposed: [o][c]
  __shared__ uint16_t Ssh[128 * 32];   // A of step5, row-major [q][p]
  __shared__ uint16_t O1T[64 * 32];    // B of step5 transposed: [o][p]
  __shared__ float red[2];
  int tid = threadIdx.x;
  int n = blockIdx.x, g = blockIdx.y;
  const float* xg = x + (((size_t)n * 4 + g) << 11);  // 32*64 = 2048
  uint16_t* pb = pbuf + (size_t)n * 32768 + (size_t)g * 8192;

  if (tid < 2) red[tid] = 0.f;
  for (int i = tid; i < 2048; i += 128) Xsh[i] = f2bf(xg[i]);
  for (int i = tid; i < 4096; i += 128) {  // M: [c][o] -> MshT[o][c]
    int cc = i >> 6, o = i & 63;
    MshT[o * 64 + cc] = pb[i];
  }
  {
    const uint32_t* ps = (const uint32_t*)(pb + 4096);
    uint32_t* ss = (uint32_t*)Ssh;
    for (int i = tid; i < 2048; i += 128) ss[i] = ps[i];
  }
  __syncthreads();

  int w = tid >> 5, L = tid & 31, h = L >> 4, ln = L & 15;

  // ---- step3: out1(32x64) = X @ M ; 8 tiles, 2 per wave
  v8f acc3[2];
#pragma unroll
  for (int e = 0; e < 2; e++) {
    int t = w * 2 + e;
    int mi = t >> 2, ni = t & 3;
    const uint32_t* arow = (const uint32_t*)(Xsh + (mi * 16 + ln) * 64);
    const uint32_t* brow = (const uint32_t*)(MshT + (ni * 16 + ln) * 64);
    v8f acc = {};
#pragma unroll
    for (int kb = 0; kb < 2; kb++) {
      v16bf a = load_frag_A(arow, kb * 16, h);
      v16bf b = load_frag_B(brow, kb * 16, h);
      acc = wmma_bf16(a, b, acc);
    }
    acc3[e] = acc;
  }
  // LayerNorm over 2048 elems (register resident)
  float s = 0.f, sq = 0.f;
#pragma unroll
  for (int e = 0; e < 2; e++)
#pragma unroll
    for (int r = 0; r < 8; r++) { float v = acc3[e][r]; s += v; sq += v * v; }
  for (int off = 16; off >= 1; off >>= 1) {
    s += __shfl_down(s, off, 32);
    sq += __shfl_down(sq, off, 32);
  }
  if (L == 0) { atomicAdd(&red[0], s); atomicAdd(&red[1], sq); }
  __syncthreads();
  float mu = red[0] * (1.f / 2048.f);
  float var = red[1] * (1.f / 2048.f) - mu * mu;
  float rstd = rsqrtf(var + 1e-5f);
  __syncthreads();
  if (tid < 2) red[tid] = 0.f;
#pragma unroll
  for (int e = 0; e < 2; e++) {
    int t = w * 2 + e;
    int mi = t >> 2, ni = t & 3;
#pragma unroll
    for (int r = 0; r < 8; r++) {
      float v = fmaxf((acc3[e][r] - mu) * rstd, 0.f);
      int p = mi * 16 + r + 8 * h;
      int o = ni * 16 + ln;
      O1T[o * 32 + p] = f2bf(v);
    }
  }
  __syncthreads();

  // ---- step5: out2(128x64) = S @ out1 ; 32 tiles, 8 per wave (K=32, one WMMA each)
  v8f acc5[2][4];
#pragma unroll
  for (int e = 0; e < 2; e++) {
    int mi = w * 2 + e;
    const uint32_t* arow = (const uint32_t*)(Ssh + (mi * 16 + ln) * 32);
    v16bf a = load_frag_A(arow, 0, h);
#pragma unroll
    for (int ni = 0; ni < 4; ni++) {
      const uint32_t* brow = (const uint32_t*)(O1T + (ni * 16 + ln) * 32);
      v16bf b = load_frag_B(brow, 0, h);
      v8f acc = {};
      acc5[e][ni] = wmma_bf16(a, b, acc);
    }
  }
  // LayerNorm over 8192 elems
  s = 0.f; sq = 0.f;
#pragma unroll
  for (int e = 0; e < 2; e++)
#pragma unroll
    for (int ni = 0; ni < 4; ni++)
#pragma unroll
      for (int r = 0; r < 8; r++) { float v = acc5[e][ni][r]; s += v; sq += v * v; }
  for (int off = 16; off >= 1; off >>= 1) {
    s += __shfl_down(s, off, 32);
    sq += __shfl_down(sq, off, 32);
  }
  if (L == 0) { atomicAdd(&red[0], s); atomicAdd(&red[1], sq); }
  __syncthreads();
  mu = red[0] * (1.f / 8192.f);
  var = red[1] * (1.f / 8192.f) - mu * mu;
  rstd = rsqrtf(var + 1e-5f);
#pragma unroll
  for (int e = 0; e < 2; e++)
#pragma unroll
    for (int ni = 0; ni < 4; ni++)
#pragma unroll
      for (int r = 0; r < 8; r++) {
        float v = fmaxf((acc5[e][ni][r] - mu) * rstd, 0.f);
        int p = (w * 2 + e) * 16 + r + 8 * h;
        int o = ni * 16 + ln;
        pb[p * 64 + o] = f2bf(v);  // overwrite consumed params region
      }
}

// ---------------- K3: out = query + out2flat(3600x32768) @ Wo(32768x256) + bo
// block: 256 thr (8 waves), tile 16x128. A tile double-buffered in LDS via
// GLOBAL_LOAD_ASYNC_TO_LDS_B128 (ASYNCcnt), B streamed from L2 with prefetch.
// grid (2, 225)
__global__ __launch_bounds__(256) void k3_proj(const uint16_t* __restrict__ pbuf,
                                               const uint32_t* __restrict__ WoT,
                                               const float* __restrict__ query,
                                               const float* __restrict__ bo,
                                               float* __restrict__ out) {
  __shared__ uint32_t As[2][16 * 128];  // 2 x 8KB: 16 rows x 512B K-chunk
  int tid = threadIdx.x;
  int n0 = blockIdx.y * 16;
  int c0 = blockIdx.x * 128;
  int w = tid >> 5, L = tid & 31, h = L >> 4, ln = L & 15;
  int c = c0 + w * 16 + ln;
  const uint32_t* brow = WoT + (size_t)c * 16384;  // 32768 bf16 K-contiguous
  const char* gbase = (const char*)pbuf;           // rows are 65536 bytes

  // issue this thread's share (2 x 16B) of an 8KB A-chunk copy, memory -> LDS
  auto issue = [&](int buf, int kc) {
    uint32_t lbase = (uint32_t)(uintptr_t)(&As[buf][0]);
#pragma unroll
    for (int it = 0; it < 2; it++) {
      int off = it * 4096 + tid * 16;          // byte offset in 8KB chunk
      int r = off >> 9, col = off & 511;       // row, byte-col within 512B
      const void* g = gbase + (((size_t)(n0 + r)) << 16) + (size_t)kc * 512 + col;
      asm volatile("global_load_async_to_lds_b128 %0, %1, off"
                   :: "v"(lbase + (uint32_t)off), "v"(g)
                   : "memory");
    }
  };

  issue(0, 0);
  v8f acc = {};
  for (int kc = 0; kc < 128; kc++) {
    asm volatile("s_wait_asynccnt 0x0" ::: "memory");  // my copies landed
    __syncthreads();                                   // everyone's landed
    if (kc + 1 < 128) issue((kc + 1) & 1, kc + 1);     // prefetch next chunk
    __builtin_prefetch(brow + (kc + 1) * 128, 0, 0);   // stream B ahead
    const uint32_t* arow = &As[kc & 1][ln * 128];
    const uint32_t* bb = brow + kc * 128;
#pragma unroll
    for (int kb = 0; kb < 8; kb++) {
      v16bf a = load_frag_A(arow, kb * 16, h);
      v16bf b = load_frag_B(bb, kb * 16, h);
      acc = wmma_bf16(a, b, acc);
    }
    __syncthreads();  // done reading As[kc&1] before it is re-issued
  }
  float bov = bo[c];
#pragma unroll
  for (int r = 0; r < 8; r++) {
    int m = r + 8 * h;
    size_t idx = (size_t)(n0 + m) * 256 + c;
    out[idx] = query[idx] + bov + acc[r];
  }
}

extern "C" void kernel_launch(void* const* d_in, const int* in_sizes, int n_in,
                              void* d_out, int out_size, void* d_ws, size_t ws_size,
                              hipStream_t stream) {
  const float* x     = (const float*)d_in[0];  // [3600,4,32,64]
  const float* query = (const float*)d_in[1];  // [3600,256]
  const float* Wp    = (const float*)d_in[2];  // [256,32768]
  const float* bp    = (const float*)d_in[3];  // [32768]
  const float* Wo    = (const float*)d_in[4];  // [32768,256]
  const float* bo    = (const float*)d_in[5];  // [256]
  float* out = (float*)d_out;

  char* ws = (char*)d_ws;
  uint16_t* WpT  = (uint16_t*)ws;                   // 32768x256 bf16 = 16 MB
  uint16_t* WoT  = (uint16_t*)(ws + 16777216);      // 256x32768 bf16 = 16 MB
  uint16_t* qbf  = (uint16_t*)(ws + 33554432);      // 3600x256 bf16
  uint16_t* pbuf = (uint16_t*)(ws + 35397632);      // 3600x32768 bf16 = 225 MB

  k0_transpose_cvt<<<dim3(1024, 8), 256, 0, stream>>>(Wp, WpT, 256, 32768);
  k0_transpose_cvt<<<dim3(8, 1024), 256, 0, stream>>>(Wo, WoT, 32768, 256);
  k0_cvt<<<3600, 256, 0, stream>>>(query, qbf, 3600 * 256);
  k1_params<<<dim3(256, 225), 256, 0, stream>>>((const uint32_t*)qbf,
                                                (const uint32_t*)WpT, bp, pbuf);
  k2_mix<<<dim3(3600, 4), 128, 0, stream>>>(x, pbuf);
  k3_proj<<<dim3(2, 225), 256, 0, stream>>>(pbuf, (const uint32_t*)WoT, query, bo, out);
}
